// SAGE_12695923327564
// MI455X (gfx1250) — matrix operand order repaired and checked by reading, not compiled, not run on hardware
//
#include <hip/hip_runtime.h>
#include <hip/hip_bf16.h>
#include <stdint.h>

// ---------------------------------------------------------------------------
// GraphSAGE 3-layer forward for MI455X (gfx1250, wave32, WMMA).
//
// Pipeline per layer:
//   zero(agg) -> scatter_kernel (edge gather + f32 atomic segment-sum)
//   sage_gemm_wmma: out = h @ W_self + (agg * invdeg) @ W_neigh  [+ bias]
//       - packed weights staged to LDS once per block (ds_load_b128 feeds WMMA)
//       - A fragments built in-register from f32 activations (cvt to f16,
//         invdeg scaling fused into the neighbor-A load)
//       - all 8 B fragments of a column tile batched into registers, then
//         8 back-to-back v_wmma_f32_16x16x32_f16 (one wait per tile)
//   bn_stats + bn_apply_relu (layers 0,1 only; in-place)
// ---------------------------------------------------------------------------

typedef __attribute__((ext_vector_type(16))) _Float16     v16h;
typedef __attribute__((ext_vector_type(8)))  float        v8f;
typedef __attribute__((ext_vector_type(4)))  float        v4f;
typedef __attribute__((ext_vector_type(4)))  unsigned int v4u;

#define HID 128
#define KCHUNKS 4              // 128 / 32
#define WAVES_PER_BLOCK 4
#define BN_ROWS 256
#define MAXPACK_V4 2048        // 16-byte chunks per packed matrix (colTiles=8)

// ---------------------------------------------------------------- utilities
__global__ void zero4_kernel(float* __restrict__ p, int n4) {
    int i = blockIdx.x * blockDim.x + threadIdx.x;
    if (i < n4) {
        v4f z = {0.f, 0.f, 0.f, 0.f};
        ((v4f*)p)[i] = z;
    }
}

__global__ void deg_kernel(const int* __restrict__ dst, float* __restrict__ deg, int E) {
    int e = blockIdx.x * blockDim.x + threadIdx.x;
    if (e < E) atomicAdd(&deg[dst[e]], 1.0f);
}

__global__ void invdeg_kernel(float* __restrict__ deg, int n) {
    int i = blockIdx.x * blockDim.x + threadIdx.x;
    if (i < n) {
        float d = deg[i];
        deg[i] = 1.0f / fmaxf(d, 1.0f);
    }
}

// ---------------------------------------------------- edge scatter (mean agg)
// 32 threads per edge, each handles a float4 chunk of the 128-wide feature.
__global__ void scatter_kernel(const float* __restrict__ h, const int* __restrict__ src,
                               const int* __restrict__ dst, float* __restrict__ agg, int E) {
    int idx = blockIdx.x * blockDim.x + threadIdx.x;
    int e = idx >> 5;
    if (e >= E) return;
    int c = (idx & 31) << 2;
    v4f v = *(const v4f*)(h + (size_t)src[e] * HID + c);
    float* o = agg + (size_t)dst[e] * HID + c;
    atomicAdd(o + 0, v[0]);
    atomicAdd(o + 1, v[1]);
    atomicAdd(o + 2, v[2]);
    atomicAdd(o + 3, v[3]);
}

// -------------------------------------------------- weight pre-pack (B frags)
// Packed layout: half index = ((ct*KCHUNKS + kc)*32 + lane)*16 + i
// B-tile (32x16, 16-bit) per ISA 7.12.2: lane n%16 holds column n,
// element i<8 -> K = half*8+i ; i>=8 -> K = 16 + half*8 + (i-8) (within the 32-chunk).
__global__ void pack_w_kernel(const float* __restrict__ W, _Float16* __restrict__ out,
                              int dout, int colTiles) {
    int idx = blockIdx.x * blockDim.x + threadIdx.x;
    int total = colTiles * KCHUNKS * 32 * 16;
    if (idx >= total) return;
    int i    = idx & 15;
    int lane = (idx >> 4) & 31;
    int kc   = (idx >> 9) & 3;
    int ct   = idx >> 11;
    int n_local = lane & 15;
    int halfSel = lane >> 4;
    int k = kc * 32 + halfSel * 8 + (i & 7) + (i >> 3) * 16;   // 0..127
    int n = ct * 16 + n_local;
    float v = (n < dout) ? W[(size_t)k * dout + n] : 0.0f;
    out[idx] = (_Float16)v;
}

// ------------------------------------------------------------- fused WMMA GEMM
// out[N x dout] = h @ Wself + (agg * invdeg) @ Wneigh  (+ bias)
// One wave per 16-row slab; loops over column tiles, A fragments resident.
__global__ __launch_bounds__(WAVES_PER_BLOCK * 32)
void sage_gemm_wmma(const float* __restrict__ h,
                    const float* __restrict__ agg,
                    const float* __restrict__ invdeg,
                    const v4u* __restrict__ pws,
                    const v4u* __restrict__ pwn,
                    float* __restrict__ out,
                    const float* __restrict__ bias,
                    int dout, int colTiles, int rowTiles) {
    __shared__ __align__(32) v4u ldsWS[MAXPACK_V4];   // 32 KB
    __shared__ __align__(32) v4u ldsWN[MAXPACK_V4];   // 32 KB

    // ---- stage packed weights to LDS (all waves participate, then barrier) ----
    const int tid = threadIdx.x;
    const int nChunks = colTiles * KCHUNKS * 32 * 2;  // 16B chunks per matrix
    for (int i = tid; i < nChunks; i += WAVES_PER_BLOCK * 32) {
        ldsWS[i] = pws[i];
        ldsWN[i] = pwn[i];
    }
    __syncthreads();

    const int lane    = threadIdx.x & 31;
    const int wave    = threadIdx.x >> 5;
    const int rowTile = blockIdx.x * WAVES_PER_BLOCK + wave;
    if (rowTile >= rowTiles) return;          // wave-uniform: EXEC stays all-1s

    const int m       = lane & 15;
    const int halfSel = lane >> 4;
    const int row     = rowTile * 16 + m;     // A-matrix row owned by this lane
    const float inv   = invdeg[row];
    const float* hrow = h   + (size_t)row * HID;
    const float* arow = agg + (size_t)row * HID;

    // Speculative prefetch of the next row-slab (emits global_prefetch_b8).
    __builtin_prefetch(hrow + (size_t)16 * WAVES_PER_BLOCK * HID, 0, 1);
    __builtin_prefetch(arow + (size_t)16 * WAVES_PER_BLOCK * HID, 0, 1);

    // Build A fragments (16x32 f16 tiles): per ISA 7.12.2, lane holds row m,
    // element i<8 -> K = half*8+i ; i>=8 -> K = 16 + half*8 + (i-8).
    v16h aS[KCHUNKS], aN[KCHUNKS];
#pragma unroll
    for (int kc = 0; kc < KCHUNKS; ++kc) {
        const int k0 = kc * 32 + halfSel * 8;
        const int k1 = k0 + 16;
        v4f x0 = *(const v4f*)(hrow + k0);
        v4f x1 = *(const v4f*)(hrow + k0 + 4);
        v4f x2 = *(const v4f*)(hrow + k1);
        v4f x3 = *(const v4f*)(hrow + k1 + 4);
        v4f y0 = *(const v4f*)(arow + k0);
        v4f y1 = *(const v4f*)(arow + k0 + 4);
        v4f y2 = *(const v4f*)(arow + k1);
        v4f y3 = *(const v4f*)(arow + k1 + 4);
#pragma unroll
        for (int i = 0; i < 4; ++i) {
            aS[kc][i]      = (_Float16)x0[i];
            aS[kc][4 + i]  = (_Float16)x1[i];
            aS[kc][8 + i]  = (_Float16)x2[i];
            aS[kc][12 + i] = (_Float16)x3[i];
            aN[kc][i]      = (_Float16)(y0[i] * inv);
            aN[kc][4 + i]  = (_Float16)(y1[i] * inv);
            aN[kc][8 + i]  = (_Float16)(y2[i] * inv);
            aN[kc][12 + i] = (_Float16)(y3[i] * inv);
        }
    }

    const v16h* lWS = (const v16h*)ldsWS;
    const v16h* lWN = (const v16h*)ldsWN;

    for (int ct = 0; ct < colTiles; ++ct) {
        // Batch all B fragments of this column tile from LDS, then WMMA chain.
        v16h bS[KCHUNKS], bN[KCHUNKS];
#pragma unroll
        for (int kc = 0; kc < KCHUNKS; ++kc) {
            bS[kc] = lWS[(ct * KCHUNKS + kc) * 32 + lane];
            bN[kc] = lWN[(ct * KCHUNKS + kc) * 32 + lane];
        }
        v8f c = {};
#pragma unroll
        for (int kc = 0; kc < KCHUNKS; ++kc) {
            c = __builtin_amdgcn_wmma_f32_16x16x32_f16(false, aS[kc], false, bS[kc],
                                                       (short)0, c, false, false);
            c = __builtin_amdgcn_wmma_f32_16x16x32_f16(false, aN[kc], false, bN[kc],
                                                       (short)0, c, false, false);
        }
        // D layout: c[r] = D[m = half*8 + r][n = lane&15]
        const int n = ct * 16 + m;
        if (n < dout) {
            const float badd = bias ? bias[n] : 0.0f;
#pragma unroll
            for (int r = 0; r < 8; ++r) {
                int orow = rowTile * 16 + halfSel * 8 + r;
                out[(size_t)orow * dout + n] = c[r] + badd;
            }
        }
    }
}

// ------------------------------------------------------------------ BatchNorm
__global__ void bn_stats_kernel(const float* __restrict__ x, float* __restrict__ stats,
                                int nrows) {
    const int c = threadIdx.x;                 // 128 threads = one per column
    const int rowBase = blockIdx.x * BN_ROWS;
    int rend = nrows - rowBase;
    if (rend > BN_ROWS) rend = BN_ROWS;
    float s = 0.f, s2 = 0.f;
    for (int r = 0; r < rend; ++r) {
        float v = x[(size_t)(rowBase + r) * HID + c];
        s  += v;
        s2 += v * v;
    }
    atomicAdd(&stats[c], s);
    atomicAdd(&stats[HID + c], s2);
}

__global__ void bn_apply_relu_kernel(float* __restrict__ x, const float* __restrict__ stats,
                                     const float* __restrict__ gamma,
                                     const float* __restrict__ beta,
                                     int total, float invN) {
    int idx = blockIdx.x * blockDim.x + threadIdx.x;
    if (idx >= total) return;
    int c = idx & (HID - 1);
    float mean = stats[c] * invN;
    float var  = stats[HID + c] * invN - mean * mean;
    float r    = rsqrtf(var + 1e-5f);
    float v    = (x[idx] - mean) * r * gamma[c] + beta[c];
    x[idx] = v > 0.f ? v : 0.f;
}

// ------------------------------------------------------------------- launcher
extern "C" void kernel_launch(void* const* d_in, const int* in_sizes, int n_in,
                              void* d_out, int out_size, void* d_ws, size_t ws_size,
                              hipStream_t stream) {
    const float* feat    = (const float*)d_in[0];
    const int*   src     = (const int*)d_in[1];
    const int*   dstp    = (const int*)d_in[2];
    const float* Wself0  = (const float*)d_in[3];
    const float* Wneigh0 = (const float*)d_in[4];
    const float* gamma0  = (const float*)d_in[5];
    const float* beta0   = (const float*)d_in[6];
    const float* Wself1  = (const float*)d_in[7];
    const float* Wneigh1 = (const float*)d_in[8];
    const float* gamma1  = (const float*)d_in[9];
    const float* beta1   = (const float*)d_in[10];
    const float* Wself2  = (const float*)d_in[11];
    const float* Wneigh2 = (const float*)d_in[12];
    const float* b2      = (const float*)d_in[13];

    const int N = in_sizes[0] / HID;   // 100000 (divisible by 16)
    const int E = in_sizes[1];         // 800000
    const int OUTD = 40;
    const int rowTiles = N / 16;
    const int ctHid = HID / 16;         // 8
    const int ctOut = (OUTD + 15) / 16; // 3 (padded to 48)

    // --- carve workspace (256B aligned slices) ---
    uintptr_t wp = (uintptr_t)d_ws;
    auto carve = [&](size_t bytes) -> void* {
        void* r = (void*)wp;
        wp += (bytes + 255) & ~(size_t)255;
        return r;
    };
    float*     invdeg = (float*)carve((size_t)N * 4);
    float*     agg    = (float*)carve((size_t)N * HID * 4);
    float*     buf1   = (float*)carve((size_t)N * HID * 4);
    float*     buf2   = (float*)carve((size_t)N * HID * 4);
    float*     stats  = (float*)carve(2 * HID * 4);
    const int packHid = ctHid * KCHUNKS * 32 * 16;  // 16384 halfs
    const int packOut = ctOut * KCHUNKS * 32 * 16;  //  6144 halfs
    _Float16* pws0 = (_Float16*)carve((size_t)packHid * 2);
    _Float16* pwn0 = (_Float16*)carve((size_t)packHid * 2);
    _Float16* pws1 = (_Float16*)carve((size_t)packHid * 2);
    _Float16* pwn1 = (_Float16*)carve((size_t)packHid * 2);
    _Float16* pws2 = (_Float16*)carve((size_t)packOut * 2);
    _Float16* pwn2 = (_Float16*)carve((size_t)packOut * 2);
    (void)ws_size; (void)n_in; (void)out_size;

    const int T = 256;
    auto cdiv = [](int a, int b) { return (a + b - 1) / b; };

    // --- degree / inverse degree (shared across layers) ---
    zero4_kernel<<<cdiv(N / 4, T), T, 0, stream>>>(invdeg, N / 4);
    deg_kernel<<<cdiv(E, T), T, 0, stream>>>(dstp, invdeg, E);
    invdeg_kernel<<<cdiv(N, T), T, 0, stream>>>(invdeg, N);

    // --- pack weights into WMMA B-fragment layout ---
    pack_w_kernel<<<cdiv(packHid, T), T, 0, stream>>>(Wself0,  pws0, HID,  ctHid);
    pack_w_kernel<<<cdiv(packHid, T), T, 0, stream>>>(Wneigh0, pwn0, HID,  ctHid);
    pack_w_kernel<<<cdiv(packHid, T), T, 0, stream>>>(Wself1,  pws1, HID,  ctHid);
    pack_w_kernel<<<cdiv(packHid, T), T, 0, stream>>>(Wneigh1, pwn1, HID,  ctHid);
    pack_w_kernel<<<cdiv(packOut, T), T, 0, stream>>>(Wself2,  pws2, OUTD, ctOut);
    pack_w_kernel<<<cdiv(packOut, T), T, 0, stream>>>(Wneigh2, pwn2, OUTD, ctOut);

    const int aggN4     = N * HID / 4;
    const int scatterB  = cdiv(E * 32, T);
    const int gemmB     = cdiv(rowTiles, WAVES_PER_BLOCK);
    const int bnStatsB  = cdiv(N, BN_ROWS);
    const int elemB     = cdiv(N * HID, T);
    const float invN    = 1.0f / (float)N;

    // ---------------- layer 0: feat(128) -> buf1(128), BN+ReLU ----------------
    zero4_kernel<<<cdiv(aggN4, T), T, 0, stream>>>(agg, aggN4);
    scatter_kernel<<<scatterB, T, 0, stream>>>(feat, src, dstp, agg, E);
    sage_gemm_wmma<<<gemmB, WAVES_PER_BLOCK * 32, 0, stream>>>(
        feat, agg, invdeg, (const v4u*)pws0, (const v4u*)pwn0,
        buf1, nullptr, HID, ctHid, rowTiles);
    zero4_kernel<<<1, 64, 0, stream>>>(stats, 64);
    bn_stats_kernel<<<bnStatsB, HID, 0, stream>>>(buf1, stats, N);
    bn_apply_relu_kernel<<<elemB, T, 0, stream>>>(buf1, stats, gamma0, beta0, N * HID, invN);

    // ---------------- layer 1: buf1(128) -> buf2(128), BN+ReLU ----------------
    zero4_kernel<<<cdiv(aggN4, T), T, 0, stream>>>(agg, aggN4);
    scatter_kernel<<<scatterB, T, 0, stream>>>(buf1, src, dstp, agg, E);
    sage_gemm_wmma<<<gemmB, WAVES_PER_BLOCK * 32, 0, stream>>>(
        buf1, agg, invdeg, (const v4u*)pws1, (const v4u*)pwn1,
        buf2, nullptr, HID, ctHid, rowTiles);
    zero4_kernel<<<1, 64, 0, stream>>>(stats, 64);
    bn_stats_kernel<<<bnStatsB, HID, 0, stream>>>(buf2, stats, N);
    bn_apply_relu_kernel<<<elemB, T, 0, stream>>>(buf2, stats, gamma1, beta1, N * HID, invN);

    // ---------------- layer 2: buf2(128) -> d_out(40), +bias ------------------
    zero4_kernel<<<cdiv(aggN4, T), T, 0, stream>>>(agg, aggN4);
    scatter_kernel<<<scatterB, T, 0, stream>>>(buf2, src, dstp, agg, E);
    sage_gemm_wmma<<<gemmB, WAVES_PER_BLOCK * 32, 0, stream>>>(
        buf2, agg, invdeg, (const v4u*)pws2, (const v4u*)pwn2,
        (float*)d_out, b2, OUTD, ctOut, rowTiles);
}
